// LocalBarrierPooling_32796370272349
// MI455X (gfx1250) — matrix operand order repaired and checked by reading, not compiled, not run on hardware
//
#include <hip/hip_runtime.h>

typedef __attribute__((ext_vector_type(16))) _Float16 v16h;
typedef __attribute__((ext_vector_type(8)))  _Float16 v8h;
typedef __attribute__((ext_vector_type(8)))  float    v8f;

#define HL      49
#define ALPHA   0.2f
#define EPSV    4.5399929762484854e-05f   /* exp(-10) */

#define Bdim    4
#define Cdim    64
#define Tdim    4096
#define TILE_T  128
#define CSPLIT  2                         /* workgroups per tile (M split)   */
#define CL      (Cdim / CSPLIT)           /* 32 channels per workgroup       */
#define NPAD    (TILE_T + 2*HL)           /* 226 padded barrier entries      */
#define XCOLS   (16*7 + 128)              /* 240: last sub-tile reads p<128  */
#define XSTR    248                       /* halves; 496B = multiple of 16B  */

__global__ __launch_bounds__(256)
void lbp_wmma_kernel(const float* __restrict__ x,
                     const float* __restrict__ barrier,
                     float* __restrict__ out)
{
    __shared__ alignas(16) _Float16 s_x[CL][XSTR];     // 15872 B
    __shared__ float s_scan[256];
    __shared__ float s_P[NPAD + 1];                    // exclusive prefix sums

    const int tid  = threadIdx.x;
    const int lane = tid & 31;
    const int wv   = tid >> 5;                         // wave id: 16-col sub-tile

    const int tilesPerB = Tdim / TILE_T;               // 32
    const int cs  = blockIdx.x & (CSPLIT - 1);         // channel-block split
    const int tb  = blockIdx.x >> 1;                   // log2(CSPLIT)==1
    const int b   = tb / tilesPerB;
    const int t0  = (tb - b * tilesPerB) * TILE_T;

    // ---------------- Phase 1: stage x slice (f16) + padded barrier ---------
    const float* xb = x + ((size_t)b * Cdim + cs * CL) * Tdim;
    for (int idx = tid; idx < CL * XCOLS; idx += 256) {
        int c  = idx / XCOLS;
        int i  = idx - c * XCOLS;
        int gt = t0 - HL + i;
        float v = (gt >= 0 && gt < Tdim) ? xb[c * Tdim + gt] : 0.0f;
        s_x[c][i] = (_Float16)v;
    }
    {
        float v = 0.0f;
        if (tid < NPAD) {
            int gt = t0 - HL + tid;
            v = (gt >= 0 && gt < Tdim) ? barrier[b * Tdim + gt] : 20.0f;
        }
        s_scan[tid] = v;                               // pad to 256 with 0
    }
    __syncthreads();

    // ---------------- Phase 2: Hillis-Steele inclusive scan -----------------
    #pragma unroll
    for (int off = 1; off < 256; off <<= 1) {
        float add = (tid >= off) ? s_scan[tid - off] : 0.0f;
        __syncthreads();
        s_scan[tid] += add;
        __syncthreads();
    }
    if (tid == 0) s_P[0] = 0.0f;
    if (tid < NPAD) s_P[tid + 1] = s_scan[tid];
    __syncthreads();

    // ---------------- Phase 3: per-lane B fragments (weights) ---------------
    // B order used here (32x16 f16, contiguous k, same as A): col n = lane&15; lanes<16 hold K=e, lanes>=16 K=16+e
    const int n    = lane & 15;
    const int hi16 = (lane >= 16) ? 1 : 0;
    const int tcol = 16 * wv + n;                      // column within tile
    const int ic   = HL + tcol;                        // center index (bl space)
    const float Pc  = s_P[ic];
    const float Pc1 = s_P[ic + 1];

    v16h  bfrag[4];
    float denp = 0.0f;
    #pragma unroll
    for (int kb = 0; kb < 4; ++kb) {
        #pragma unroll
        for (int e = 0; e < 16; ++e) {
            int p = 32 * kb + e + (hi16 ? 16 : 0);     // K index 0..127
            int d = p - HL - n;                        // tap offset
            float wt = 0.0f;
            if (d >= -HL && d <= HL) {
                float s;
                if (d > 0)      s = s_P[ic + d + 1] - Pc;   // center..t+d
                else if (d < 0) s = Pc1 - s_P[ic + d];      // t+d..center
                else            s = 0.0f;
                wt = __expf(-ALPHA * s);
            }
            _Float16 h = (_Float16)wt;
            bfrag[kb][e] = h;
            denp += (float)h;
        }
    }
    // Lanes L and L^16 hold disjoint halves of K for the same column.
    denp += __shfl_xor(denp, 16, 32);
    const float inv = 1.0f / (denp + EPSV);

    // ---------------- Phase 4: 32x128x16 GEMM via 8 WMMAs -------------------
    // A order used here (16x32 f16, contiguous k, same as B): row m = lane&15; lanes<16: K [0,16);
    // lanes>=16: K [16,32).
    const int alo = hi16 ? 16 : 0;
    float* outb = out + ((size_t)b * Cdim + cs * CL) * Tdim + (t0 + 16 * wv);

    #pragma unroll
    for (int mb = 0; mb < CL / 16; ++mb) {
        v8f acc = {};
        const int ca = 16 * mb + n;                    // A row = lane&15
        #pragma unroll
        for (int kb = 0; kb < 4; ++kb) {
            const int coff = 16 * wv + 32 * kb;
            const v8h a0 = *(const v8h*)&s_x[ca][coff + alo];
            const v8h a1 = *(const v8h*)&s_x[ca][coff + alo + 8];
            v16h a;
            #pragma unroll
            for (int e = 0; e < 8; ++e) { a[e] = a0[e]; a[e + 8] = a1[e]; }
            acc = __builtin_amdgcn_wmma_f32_16x16x32_f16(
                      false, a, false, bfrag[kb], (short)0, acc, false, false);
        }
        // C/D layout: row = r + 8*(lane>=16), col = lane&15
        const int crow = 16 * mb + (hi16 ? 8 : 0);
        #pragma unroll
        for (int r = 0; r < 8; ++r) {
            outb[(size_t)(crow + r) * Tdim + n] = acc[r] * inv;
        }
    }
}

extern "C" void kernel_launch(void* const* d_in, const int* in_sizes, int n_in,
                              void* d_out, int out_size, void* d_ws, size_t ws_size,
                              hipStream_t stream)
{
    const float* xp   = (const float*)d_in[0];
    const float* barp = (const float*)d_in[1];
    float* outp = (float*)d_out;
    dim3 grid(Bdim * (Tdim / TILE_T) * CSPLIT);  // 4 * 32 * 2 = 256 workgroups
    dim3 block(256);                             // 8 waves; one sub-tile each
    lbp_wmma_kernel<<<grid, block, 0, stream>>>(xp, barp, outp);
}
